// get_uncertainty_12515534701189
// MI455X (gfx1250) — matrix operand order repaired and checked by reading, not compiled, not run on hardware
//
#include <hip/hip_runtime.h>

// ---------------- problem constants ----------------
#define Bdim   8
#define Cdim   256
#define HWdim  16384                  // 128*128
#define NPIX   (Bdim * HWdim)         // 131072
#define CHW    ((size_t)Cdim * HWdim) // 4194304

// ---------------- workspace layout (bytes) ----------------
static const size_t WS_HPRE   = 0;          // 134,217,728  h_pre f32
static const size_t WS_U0     = 134217728;  // 524,288
static const size_t WS_U1     = 134742016;  // 524,288
static const size_t WS_WP     = 135266304;  // 131,072 bf16 W_proj
static const size_t WS_WC     = 135397376;  // 131,072 bf16 W_conv
static const size_t WS_SUM    = 135528448;
static const size_t WS_SQ     = 135529472;
static const size_t WS_SCALE  = 135530496;
static const size_t WS_SHIFT  = 135531520;
static const size_t WS_MINMAX = 135532544;

// ------------- dynamic-LDS layout for the GEMM kernels (bytes) -------------
#define ROWA      264                      // 256 + 8 pad; row stride 528 B (16B aligned)
#define A_BYTES   (128 * ROWA * 2)         // 67,584  : full 128x256 bf16 A half
#define ROWB      40                       // 32 + 8 pad; row stride 80 B (16B aligned)
#define B_BYTES   (2 * 64 * ROWB * 2)      // 10,240  : double-buffered 64x32 bf16 B
#define SMEM_GEMM (A_BYTES + B_BYTES + 2048) // 79,872

typedef __attribute__((ext_vector_type(16))) __bf16 bf16x16;
typedef __attribute__((ext_vector_type(8)))  float  f32x8;

union FragBF { unsigned int u[8]; uint4 q[2]; bf16x16 v; };

__device__ __forceinline__ unsigned short f2bf(float f) {
    unsigned int x = __float_as_uint(f);
    x += 0x7FFFu + ((x >> 16) & 1u);          // round-to-nearest-even
    return (unsigned short)(x >> 16);
}

// RNE-round two f32 to bf16 and pack (lo = a, hi = b) with one v_perm_b32.
__device__ __forceinline__ unsigned int pack2_bf16(float a, float b) {
    unsigned int xa = __float_as_uint(a);
    unsigned int xb = __float_as_uint(b);
    xa += 0x7FFFu + ((xa >> 16) & 1u);
    xb += 0x7FFFu + ((xb >> 16) & 1u);
    return __builtin_amdgcn_perm(xb, xa, 0x07060302u);
}

// A fragment, 16-bit A 16x32 layout (ISA 7.12.2): lane -> row m=l&15, half=l>>4;
// v0..3 = K {half*8+0..7}, v4..7 = K {16+half*8+0..7}. Both quads are 16B-aligned
// contiguous in LDS -> two ds_load_b128.
__device__ __forceinline__ void load_fragA(FragBF& f, const unsigned short* row) {
    const int half = (threadIdx.x & 31) >> 4;
    f.q[0] = *(const uint4*)(row + half * 8);
    f.q[1] = *(const uint4*)(row + 16 + half * 8);
}

// B fragment matching A's K order: lane -> column n=l&15, half=l>>4;
// v0..3 = K {half*8+0..7}, v4..7 = K {16+half*8+0..7} -> two ds_load_b128.
__device__ __forceinline__ void load_fragB(FragBF& f, const unsigned short* row) {
    const int half = (threadIdx.x & 31) >> 4;
    f.q[0] = *(const uint4*)(row + half * 8);
    f.q[1] = *(const uint4*)(row + 16 + half * 8);
}

// ---------------- small setup kernels ----------------
__global__ void init_ws(float* gsum, float* gsq, unsigned int* mm) {
    const int t = threadIdx.x;
    gsum[t] = 0.f; gsq[t] = 0.f;
    if (t == 0) { mm[0] = 0x7F800000u; mm[1] = 0u; }
}

__global__ void cvt_weights(const float* __restrict__ Wp, const float* __restrict__ Wc,
                            unsigned short* __restrict__ Wpb, unsigned short* __restrict__ Wcb) {
    const int i = blockIdx.x * 256 + threadIdx.x;   // 65536 total
    Wpb[i] = f2bf(Wp[i]);
    Wcb[i] = f2bf(Wc[i]);
}

__global__ void bn_finalize(const float* __restrict__ gsum, const float* __restrict__ gsq,
                            const float* __restrict__ gamma, const float* __restrict__ beta,
                            float* __restrict__ scale, float* __restrict__ shift) {
    const int c = threadIdx.x;
    const float invN = 1.0f / (float)NPIX;
    const float mu  = gsum[c] * invN;
    const float var = gsq[c] * invN - mu * mu;
    const float rs  = rsqrtf(var + 1e-5f) * gamma[c];
    scale[c] = rs;
    shift[c] = beta[c] - mu * rs;
}

// ---------------- GEMM1: h_pre = W_proj * x, + BN partial sums ----------------
// grid = (256, 2, 8): x -> 64-pixel strip, y -> 128-channel half, z -> batch
__global__ __launch_bounds__(256) void gemm1_proj(
    const float* __restrict__ x, const unsigned short* __restrict__ Wp,
    float* __restrict__ hpre, float* __restrict__ gsum, float* __restrict__ gsq)
{
    extern __shared__ char smem[];
    unsigned short* As = (unsigned short*)smem;                  // [128][ROWA]
    unsigned short* Bs = (unsigned short*)(smem + A_BYTES);      // [2][64][ROWB]
    float* ls = (float*)(smem + A_BYTES + B_BYTES);              // [128]
    float* lq = ls + 128;

    const int t    = threadIdx.x;
    const int lane = t & 31, wave = t >> 5;
    const int b    = blockIdx.z;
    const int m0   = blockIdx.y * 128;
    const int p0   = blockIdx.x * 64;
    const float* xb = x + (size_t)b * CHW + p0;

    if (t < 128) { ls[t] = 0.f; lq[t] = 0.f; }

    { // load the full 128x256 bf16 A half into LDS once (64 KB)
        const int r = t >> 1, hc = (t & 1) * 128;
        const unsigned short* src = Wp + (m0 + r) * 256 + hc;
        unsigned short* dst = As + r * ROWA + hc;
#pragma unroll
        for (int j = 0; j < 8; ++j)
            *(uint4*)(dst + 16 * j) = *(const uint4*)(src + 16 * j);
    }

    // B staging: thread -> K-pair (kr, kr+1) x 4 pixels; pack pairs, 4x ds_store_b32
    const int kr = (t >> 4) * 2;
    const int pc = (t & 15) * 4;
    {
        const float* s0 = xb + (size_t)kr * HWdim + pc;
        float4 f0 = *(const float4*)s0;
        float4 f1 = *(const float4*)(s0 + HWdim);
        unsigned short* brow = Bs + pc * ROWB + kr;
        *(unsigned int*)(brow)            = pack2_bf16(f0.x, f1.x);
        *(unsigned int*)(brow + ROWB)     = pack2_bf16(f0.y, f1.y);
        *(unsigned int*)(brow + 2 * ROWB) = pack2_bf16(f0.z, f1.z);
        *(unsigned int*)(brow + 3 * ROWB) = pack2_bf16(f0.w, f1.w);
    }
    __syncthreads();

    const f32x8 z8 = {0.f,0.f,0.f,0.f,0.f,0.f,0.f,0.f};
    f32x8 acc[4] = { z8, z8, z8, z8 };
    const unsigned short* arow = As + (wave * 16 + (lane & 15)) * ROWA;

    for (int ks = 0; ks < 8; ++ks) {
        if (ks < 7) { // stage next K-slab into the other buffer (overlaps WMMA)
            const float* s0 = xb + (size_t)((ks + 1) * 32 + kr) * HWdim + pc;
            float4 f0 = *(const float4*)s0;
            float4 f1 = *(const float4*)(s0 + HWdim);
            unsigned short* brow = Bs + ((ks + 1) & 1) * 64 * ROWB + pc * ROWB + kr;
            *(unsigned int*)(brow)            = pack2_bf16(f0.x, f1.x);
            *(unsigned int*)(brow + ROWB)     = pack2_bf16(f0.y, f1.y);
            *(unsigned int*)(brow + 2 * ROWB) = pack2_bf16(f0.z, f1.z);
            *(unsigned int*)(brow + 3 * ROWB) = pack2_bf16(f0.w, f1.w);
        }
        const unsigned short* bbase = Bs + (ks & 1) * 64 * ROWB;
        FragBF fa;
        load_fragA(fa, arow + ks * 32);
#pragma unroll
        for (int nt = 0; nt < 4; ++nt) {
            FragBF fb;
            load_fragB(fb, bbase + (nt * 16 + (lane & 15)) * ROWB);
            acc[nt] = __builtin_amdgcn_wmma_f32_16x16x32_bf16(
                false, fa.v, false, fb.v, (short)0, acc[nt], false, false);
        }
        __syncthreads();
    }

    // write h_pre + per-channel partial sums (C/D layout: m_rel = i + 8*(lane>>4))
    const int half = lane >> 4;
    const int ncol = lane & 15;
#pragma unroll
    for (int nt = 0; nt < 4; ++nt) {
#pragma unroll
        for (int i = 0; i < 8; ++i) {
            const int mloc = wave * 16 + i + 8 * half;
            const float v = acc[nt][i];
            hpre[(size_t)b * CHW + (size_t)(m0 + mloc) * HWdim + p0 + nt * 16 + ncol] = v;
            atomicAdd(&ls[mloc], v);
            atomicAdd(&lq[mloc], v * v);
        }
    }
    __syncthreads();
    if (t < 128) {
        atomicAdd(&gsum[m0 + t], ls[t]);
        atomicAdd(&gsq[m0 + t],  lq[t]);
    }
}

// ------- GEMM2: residual = W_conv * relu(bn(h_pre)) + b_conv -> d_out -------
__global__ __launch_bounds__(256) void gemm2_resid(
    const float* __restrict__ hpre, const unsigned short* __restrict__ Wc,
    const float* __restrict__ scale, const float* __restrict__ shift,
    const float* __restrict__ bconv, float* __restrict__ resid)
{
    extern __shared__ char smem[];
    unsigned short* As = (unsigned short*)smem;
    unsigned short* Bs = (unsigned short*)(smem + A_BYTES);
    float* lscale = (float*)(smem + A_BYTES + B_BYTES);          // [256]
    float* lshift = lscale + 256;

    const int t    = threadIdx.x;
    const int lane = t & 31, wave = t >> 5;
    const int b    = blockIdx.z;
    const int m0   = blockIdx.y * 128;
    const int p0   = blockIdx.x * 64;
    const float* hb = hpre + (size_t)b * CHW + p0;

    lscale[t] = scale[t]; lshift[t] = shift[t];

    {
        const int r = t >> 1, hc = (t & 1) * 128;
        const unsigned short* src = Wc + (m0 + r) * 256 + hc;
        unsigned short* dst = As + r * ROWA + hc;
#pragma unroll
        for (int j = 0; j < 8; ++j)
            *(uint4*)(dst + 16 * j) = *(const uint4*)(src + 16 * j);
    }
    __syncthreads();   // lscale/lshift + As visible

    const int kr = (t >> 4) * 2;
    const int pc = (t & 15) * 4;
    { // first B slab, with BN+ReLU fused into the bf16 conversion
        const float sc0 = lscale[kr],     sh0 = lshift[kr];
        const float sc1 = lscale[kr + 1], sh1 = lshift[kr + 1];
        const float* s0 = hb + (size_t)kr * HWdim + pc;
        float4 f0 = *(const float4*)s0;
        float4 f1 = *(const float4*)(s0 + HWdim);
        unsigned short* brow = Bs + pc * ROWB + kr;
        *(unsigned int*)(brow)            = pack2_bf16(fmaxf(fmaf(f0.x, sc0, sh0), 0.f), fmaxf(fmaf(f1.x, sc1, sh1), 0.f));
        *(unsigned int*)(brow + ROWB)     = pack2_bf16(fmaxf(fmaf(f0.y, sc0, sh0), 0.f), fmaxf(fmaf(f1.y, sc1, sh1), 0.f));
        *(unsigned int*)(brow + 2 * ROWB) = pack2_bf16(fmaxf(fmaf(f0.z, sc0, sh0), 0.f), fmaxf(fmaf(f1.z, sc1, sh1), 0.f));
        *(unsigned int*)(brow + 3 * ROWB) = pack2_bf16(fmaxf(fmaf(f0.w, sc0, sh0), 0.f), fmaxf(fmaf(f1.w, sc1, sh1), 0.f));
    }
    __syncthreads();

    const f32x8 z8 = {0.f,0.f,0.f,0.f,0.f,0.f,0.f,0.f};
    f32x8 acc[4] = { z8, z8, z8, z8 };
    const unsigned short* arow = As + (wave * 16 + (lane & 15)) * ROWA;

    for (int ks = 0; ks < 8; ++ks) {
        if (ks < 7) {
            const int kg = (ks + 1) * 32 + kr;
            const float sc0 = lscale[kg],     sh0 = lshift[kg];
            const float sc1 = lscale[kg + 1], sh1 = lshift[kg + 1];
            const float* s0 = hb + (size_t)kg * HWdim + pc;
            float4 f0 = *(const float4*)s0;
            float4 f1 = *(const float4*)(s0 + HWdim);
            unsigned short* brow = Bs + ((ks + 1) & 1) * 64 * ROWB + pc * ROWB + kr;
            *(unsigned int*)(brow)            = pack2_bf16(fmaxf(fmaf(f0.x, sc0, sh0), 0.f), fmaxf(fmaf(f1.x, sc1, sh1), 0.f));
            *(unsigned int*)(brow + ROWB)     = pack2_bf16(fmaxf(fmaf(f0.y, sc0, sh0), 0.f), fmaxf(fmaf(f1.y, sc1, sh1), 0.f));
            *(unsigned int*)(brow + 2 * ROWB) = pack2_bf16(fmaxf(fmaf(f0.z, sc0, sh0), 0.f), fmaxf(fmaf(f1.z, sc1, sh1), 0.f));
            *(unsigned int*)(brow + 3 * ROWB) = pack2_bf16(fmaxf(fmaf(f0.w, sc0, sh0), 0.f), fmaxf(fmaf(f1.w, sc1, sh1), 0.f));
        }
        const unsigned short* bbase = Bs + (ks & 1) * 64 * ROWB;
        FragBF fa;
        load_fragA(fa, arow + ks * 32);
#pragma unroll
        for (int nt = 0; nt < 4; ++nt) {
            FragBF fb;
            load_fragB(fb, bbase + (nt * 16 + (lane & 15)) * ROWB);
            acc[nt] = __builtin_amdgcn_wmma_f32_16x16x32_bf16(
                false, fa.v, false, fb.v, (short)0, acc[nt], false, false);
        }
        __syncthreads();
    }

    const int half = lane >> 4;
    const int ncol = lane & 15;
#pragma unroll
    for (int nt = 0; nt < 4; ++nt) {
#pragma unroll
        for (int i = 0; i < 8; ++i) {
            const int mloc = wave * 16 + i + 8 * half;
            resid[(size_t)b * CHW + (size_t)(m0 + mloc) * HWdim + p0 + nt * 16 + ncol] =
                acc[nt][i] + bconv[m0 + mloc];
        }
    }
}

// ---- per-pixel: mean/logvar dots, std, prob_x, 50-sample sigmoid variance ----
__global__ __launch_bounds__(256) void pixel_stats(
    const float* __restrict__ hpre, const float* __restrict__ scale,
    const float* __restrict__ shift, const float* __restrict__ wmean,
    const float* __restrict__ wstd, const float* __restrict__ eps1,
    const float* __restrict__ eps50, float* __restrict__ probx,
    float* __restrict__ u0)
{
    __shared__ float pm[8][32], ps[8][32];
    __shared__ float ldsM[32], ldsS[32];
    __shared__ float lsc[256], lsh[256], lwm[256], lws[256];

    const int t  = threadIdx.x;
    const int p  = t & 31, g = t >> 5;
    const int b  = blockIdx.x >> 9;
    const int p0 = (blockIdx.x & 511) * 32;
    const size_t hb = (size_t)b * CHW;

    lsc[t] = scale[t]; lsh[t] = shift[t]; lwm[t] = wmean[t]; lws[t] = wstd[t];
    __syncthreads();

    float am = 0.f, as = 0.f;
#pragma unroll 4
    for (int k = 0; k < 32; ++k) {
        const int c = g + (k << 3);
        const float h = hpre[hb + (size_t)c * HWdim + p0 + p];
        const float v = fmaxf(fmaf(h, lsc[c], lsh[c]), 0.f);
        am = fmaf(lwm[c], v, am);
        as = fmaf(lws[c], v, as);
    }
    pm[g][p] = am; ps[g][p] = as;
    __syncthreads();
    if (t < 32) {
        float m = 0.f, lv = 0.f;
#pragma unroll
        for (int gg = 0; gg < 8; ++gg) { m += pm[gg][t]; lv += ps[gg][t]; }
        const float sd = __expf(0.5f * lv);
        ldsM[t] = m; ldsS[t] = sd;
        const int pp = b * HWdim + p0 + t;
        probx[pp] = fmaf(eps1[pp], sd, m);
    }
    __syncthreads();
    const float m  = ldsM[p];
    const float sd = ldsS[p];
    const size_t eb = (size_t)b * 50 * HWdim + p0 + p;
    float s1 = 0.f, s2 = 0.f;
    for (int i = g; i < 50; i += 8) {
        const float e   = eps50[eb + (size_t)i * HWdim];
        const float zz  = fmaf(e, sd, m);
        const float sig = 1.0f / (1.0f + __expf(-zz));
        s1 += sig; s2 = fmaf(sig, sig, s2);
    }
    __syncthreads();
    pm[g][p] = s1; ps[g][p] = s2;
    __syncthreads();
    if (t < 32) {
        float S1 = 0.f, S2 = 0.f;
#pragma unroll
        for (int gg = 0; gg < 8; ++gg) { S1 += pm[gg][t]; S2 += ps[gg][t]; }
        const float u = (S2 - S1 * S1 * (1.0f / 50.0f)) * (1.0f / 49.0f);
        u0[b * HWdim + p0 + t] = fmaxf(u, 0.f);   // >=0 keeps uint min/max valid
    }
}

// --------- one 7x7 zero-padded box-filter pass over [8,128,128] ---------
__global__ __launch_bounds__(256) void box7(const float* __restrict__ in,
                                            float* __restrict__ out)
{
    const int idx = blockIdx.x * 256 + threadIdx.x;   // 0..131071
    const int b = idx >> 14;
    const int pix = idx & 16383;
    const int y = pix >> 7, xc = pix & 127;
    const float* base = in + (b << 14);
    float s = 0.f;
    for (int dy = -3; dy <= 3; ++dy) {
        const int yy = y + dy;
        if (yy < 0 || yy > 127) continue;
        for (int dx = -3; dx <= 3; ++dx) {
            const int xx = xc + dx;
            if (xx < 0 || xx > 127) continue;
            s += base[(yy << 7) + xx];
        }
    }
    out[idx] = s;
}

// --------- global min/max of non-negative u via uint-bit atomics ---------
__global__ __launch_bounds__(256) void minmax_k(const float* __restrict__ u,
                                                unsigned int* __restrict__ mm)
{
    __shared__ unsigned int smn[256], smx[256];
    const int idx = blockIdx.x * 256 + threadIdx.x;
    const unsigned int bits = __float_as_uint(u[idx]);
    smn[threadIdx.x] = bits; smx[threadIdx.x] = bits;
    __syncthreads();
    for (int s = 128; s > 0; s >>= 1) {
        if (threadIdx.x < s) {
            unsigned int a = smn[threadIdx.x], b2 = smn[threadIdx.x + s];
            smn[threadIdx.x] = (a < b2) ? a : b2;
            a = smx[threadIdx.x]; b2 = smx[threadIdx.x + s];
            smx[threadIdx.x] = (a > b2) ? a : b2;
        }
        __syncthreads();
    }
    if (threadIdx.x == 0) {
        atomicMin(&mm[0], smn[0]);
        atomicMax(&mm[1], smx[0]);
    }
}

// --------- final: residual *= (1-u_norm) * (u_norm < rand_u) ---------
__global__ __launch_bounds__(256) void final_mask(
    float* __restrict__ resid, const float* __restrict__ u,
    const float* __restrict__ randu, const unsigned int* __restrict__ mm)
{
    const size_t idx = (size_t)blockIdx.x * 256 + threadIdx.x;
    const float mn = __uint_as_float(mm[0]);
    const float mx = __uint_as_float(mm[1]);
    const float inv = 1.0f / (mx - mn);
    const size_t up = ((idx >> 22) << 14) + (idx & 16383);  // b*HW + pixel
    const float un = (u[up] - mn) * inv;
    float r = resid[idx];
    r *= (1.0f - un);
    r *= (un < randu[up]) ? 1.0f : 0.0f;
    resid[idx] = r;
}

// ---------------- host-side orchestration ----------------
extern "C" void kernel_launch(void* const* d_in, const int* in_sizes, int n_in,
                              void* d_out, int out_size, void* d_ws, size_t ws_size,
                              hipStream_t stream) {
    const float* x     = (const float*)d_in[0];
    const float* Wproj = (const float*)d_in[1];
    const float* gamma = (const float*)d_in[2];
    const float* beta  = (const float*)d_in[3];
    const float* Wconv = (const float*)d_in[4];
    const float* bconv = (const float*)d_in[5];
    const float* Wmean = (const float*)d_in[6];
    const float* Wstd  = (const float*)d_in[7];
    const float* eps1  = (const float*)d_in[8];
    const float* eps50 = (const float*)d_in[9];
    const float* randu = (const float*)d_in[10];

    float* out_resid = (float*)d_out;                       // [8,256,128,128]
    float* out_probx = out_resid + (size_t)Bdim * CHW;      // [8,1,128,128]

    char* ws = (char*)d_ws;
    float*          hpre  = (float*)(ws + WS_HPRE);
    float*          u0    = (float*)(ws + WS_U0);
    float*          u1    = (float*)(ws + WS_U1);
    unsigned short* Wpb   = (unsigned short*)(ws + WS_WP);
    unsigned short* Wcb   = (unsigned short*)(ws + WS_WC);
    float*          gsum  = (float*)(ws + WS_SUM);
    float*          gsq   = (float*)(ws + WS_SQ);
    float*          scale = (float*)(ws + WS_SCALE);
    float*          shift = (float*)(ws + WS_SHIFT);
    unsigned int*   mm    = (unsigned int*)(ws + WS_MINMAX);

    init_ws<<<1, 256, 0, stream>>>(gsum, gsq, mm);
    cvt_weights<<<256, 256, 0, stream>>>(Wproj, Wconv, Wpb, Wcb);
    gemm1_proj<<<dim3(256, 2, 8), 256, SMEM_GEMM, stream>>>(x, Wpb, hpre, gsum, gsq);
    bn_finalize<<<1, 256, 0, stream>>>(gsum, gsq, gamma, beta, scale, shift);
    gemm2_resid<<<dim3(256, 2, 8), 256, SMEM_GEMM, stream>>>(hpre, Wcb, scale, shift,
                                                             bconv, out_resid);
    pixel_stats<<<4096, 256, 0, stream>>>(hpre, scale, shift, Wmean, Wstd,
                                          eps1, eps50, out_probx, u0);
    box7<<<512, 256, 0, stream>>>(u0, u1);
    box7<<<512, 256, 0, stream>>>(u1, u0);
    box7<<<512, 256, 0, stream>>>(u0, u1);
    minmax_k<<<512, 256, 0, stream>>>(u1, mm);
    final_mask<<<131072, 256, 0, stream>>>(out_resid, u1, randu, mm);
}
